// FlowGuidedDA_44040594653602
// MI455X (gfx1250) — compile-verified, matrix operand order checked by reading, not run on hardware
//
#include <hip/hip_runtime.h>
#include <hip/hip_bf16.h>
#include <math.h>

// ---------------------------------------------------------------------------
// FlowGuidedDA on MI455X (gfx1250): bf16 WMMA implicit-GEMM convs with
// TDM (tensor_load_to_lds) staging of activation strips + fused gather-GEMM
// deformable convs.  wave32, v_wmma_f32_16x16x32_bf16, f32 accumulate.
// Conv inputs live in zero-padded NHWC (halo) so the GEMM inner loop has no
// boundary masking at all.
// ---------------------------------------------------------------------------

typedef __attribute__((ext_vector_type(16))) __bf16 v16bf;
typedef __attribute__((ext_vector_type(8)))  float  v8f;
typedef __attribute__((ext_vector_type(4)))  unsigned int u32x4;
typedef __attribute__((ext_vector_type(8)))  int          i32x8;
typedef __attribute__((ext_vector_type(4)))  int          i32x4;

static constexpr int kB   = 2;
static constexpr int kC   = 64;
static constexpr int kH   = 128;
static constexpr int kW   = 256;
static constexpr int kHW  = kH * kW;          // 32768
static constexpr int kP   = kB * kHW;         // 65536 pixels
static constexpr int kPH  = kH + 2;           // padded height 130
static constexpr int kPW  = kW + 2;           // padded width  258
static constexpr int kPP  = kPH * kPW;        // 33540 padded pixels / image
static constexpr float kMRM = 20.0f;

// ---------------------------------------------------------------------------
// TDM: load a [rows x widthElems] bf16 strip (row stride rowStrideElems) from
// global into LDS at byte offset ldsOff.  D# packing per CDNA5 ISA ch.8.
// This toolchain's builtin is the 6-arg form:
//   (u32x4 group0, i32x8 group1, i32x4 group2, i32x4 group3, i32x8, i32 cpol)
// ---------------------------------------------------------------------------
__device__ __forceinline__ void tdm_load_strip(const __bf16* gptr, unsigned ldsOff,
                                               int widthElems, int rows,
                                               long rowStrideElems) {
#if __has_builtin(__builtin_amdgcn_tensor_load_to_lds)
  unsigned long long ga = (unsigned long long)(size_t)gptr;
  u32x4 g0;
  g0[0] = 1u;                                   // count=1, user descriptor
  g0[1] = ldsOff;                               // lds_addr
  g0[2] = (unsigned)(ga & 0xFFFFFFFFu);         // global_addr[31:0]
  g0[3] = (unsigned)((ga >> 32) & 0x01FFFFFFu)  // global_addr[56:32]
          | (2u << 30);                         // type = 2 (image)
  unsigned td0 = 0x7FFFFFFFu;                   // tensor_dim0 (huge: no OOB)
  unsigned td1 = 0x7FFFFFFFu;                   // tensor_dim1
  unsigned long long s0 = (unsigned long long)rowStrideElems;
  i32x8 g1;
  g1[0] = 0x10000;                              // wg_mask=0, data_size=1 (2B)
  g1[1] = (int)((td0 & 0xFFFFu) << 16);         // tensor_dim0[15:0]
  g1[2] = (int)((td0 >> 16) | ((td1 & 0xFFFFu) << 16));
  g1[3] = (int)((td1 >> 16) | ((unsigned)widthElems << 16));  // tile_dim0
  g1[4] = rows & 0xFFFF;                        // tile_dim1 = rows, tile_dim2=0
  g1[5] = (int)(s0 & 0xFFFFFFFFu);              // tensor_dim0_stride[31:0]
  g1[6] = (int)((s0 >> 32) & 0xFFFFu);          // stride[47:32], dim1_stride=0
  g1[7] = 0;
  i32x4 gz;
  gz[0] = 0; gz[1] = 0; gz[2] = 0; gz[3] = 0;
  i32x8 gz8;
  gz8[0] = 0; gz8[1] = 0; gz8[2] = 0; gz8[3] = 0;
  gz8[4] = 0; gz8[5] = 0; gz8[6] = 0; gz8[7] = 0;
  __builtin_amdgcn_tensor_load_to_lds(g0, g1, gz, gz, gz8, 0);
  __builtin_amdgcn_s_wait_tensorcnt(0);
#else
  (void)gptr; (void)ldsOff; (void)widthElems; (void)rows; (void)rowStrideElems;
#endif
}

// ---------------------------------------------------------------------------
// Weight reorder: w[Cout][Cin][3][3] f32 -> wr[9][CoutPad][CinPad] bf16 (0-pad)
// ---------------------------------------------------------------------------
__global__ __launch_bounds__(256) void prep_conv_w(const float* __restrict__ w,
                                                   __bf16* __restrict__ wr,
                                                   int Cout, int CoutPad,
                                                   int Cin, int CinPad) {
  long i = (long)blockIdx.x * 256 + threadIdx.x;
  long tot = (long)9 * CoutPad * CinPad;
  if (i >= tot) return;
  int ci = (int)(i % CinPad);
  long r = i / CinPad;
  int co = (int)(r % CoutPad);
  int t  = (int)(r / CoutPad);
  float v = 0.0f;
  if (co < Cout && ci < Cin) v = w[((size_t)co * Cin + ci) * 9 + t];
  wr[i] = (__bf16)v;
}

// Deform weights: w[O=64][C=64][9] -> wr[o][k*64 + c] bf16
__global__ __launch_bounds__(256) void prep_dc_w(const float* __restrict__ w,
                                                 __bf16* __restrict__ wr) {
  int i = blockIdx.x * 256 + threadIdx.x;   // 64*576
  if (i >= 64 * 576) return;
  int c = i % 64;
  int r = i / 64;
  int k = r % 9;
  int o = r / 9;
  wr[(size_t)o * 576 + k * 64 + c] = (__bf16)w[((size_t)o * 64 + c) * 9 + k];
}

// Zero the halo ring of a padded NHWC buffer (stride S channels per pixel).
__global__ __launch_bounds__(256) void zero_halo(__bf16* __restrict__ buf, int S) {
  int i = blockIdx.x * 256 + threadIdx.x;    // over kB * kPP padded pixels
  if (i >= kB * kPP) return;
  int pp = i % kPP;
  int y = pp / kPW, x = pp % kPW;
  if (y != 0 && y != kPH - 1 && x != 0 && x != kPW - 1) return;
  __bf16* o = buf + (size_t)i * S;
  for (int c = 0; c < S; ++c) o[c] = (__bf16)0.f;
}

// ---------------------------------------------------------------------------
// flow_warp(non_ref, flow) + concat [warped(64)|ref(64)|flow(2)|pad(30)]
// -> hin padded NHWC bf16, C=160
// ---------------------------------------------------------------------------
__global__ __launch_bounds__(256) void warp_build(const float* __restrict__ nonref,
                                                  const float* __restrict__ ref,
                                                  const float* __restrict__ flow,
                                                  __bf16* __restrict__ hin) {
  int p = blockIdx.x * 256 + threadIdx.x;
  if (p >= kP) return;
  int b  = p >> 15;
  int sp = p & (kHW - 1);
  int y  = sp >> 8;
  int x  = sp & (kW - 1);

  float fx = flow[((size_t)(b * 2 + 0)) * kHW + sp];
  float fy = flow[((size_t)(b * 2 + 1)) * kHW + sp];
  float py = (float)y + fy;
  float px = (float)x + fx;
  float y0f = floorf(py), x0f = floorf(px);
  int   yi = (int)y0f,    xi = (int)x0f;
  float wy = py - y0f,    wx = px - x0f;
  float w00 = (1.f - wy) * (1.f - wx);
  float w01 = (1.f - wy) * wx;
  float w10 = wy * (1.f - wx);
  float w11 = wy * wx;
  bool v00 = (yi >= 0)     && (yi < kH)     && (xi >= 0)     && (xi < kW);
  bool v01 = (yi >= 0)     && (yi < kH)     && (xi + 1 >= 0) && (xi + 1 < kW);
  bool v10 = (yi + 1 >= 0) && (yi + 1 < kH) && (xi >= 0)     && (xi < kW);
  bool v11 = (yi + 1 >= 0) && (yi + 1 < kH) && (xi + 1 >= 0) && (xi + 1 < kW);
  int yc0 = min(max(yi, 0), kH - 1),     yc1 = min(max(yi + 1, 0), kH - 1);
  int xc0 = min(max(xi, 0), kW - 1),     xc1 = min(max(xi + 1, 0), kW - 1);
  int i00 = yc0 * kW + xc0, i01 = yc0 * kW + xc1;
  int i10 = yc1 * kW + xc0, i11 = yc1 * kW + xc1;

  __bf16* o = hin + ((size_t)(b * kPH + y + 1) * kPW + x + 1) * 160;
  for (int c = 0; c < kC; ++c) {
    const float* base = nonref + ((size_t)(b * kC + c)) * kHW;
    float acc = 0.f;
    if (v00) acc += w00 * base[i00];
    if (v01) acc += w01 * base[i01];
    if (v10) acc += w10 * base[i10];
    if (v11) acc += w11 * base[i11];
    o[c]      = (__bf16)acc;
    o[64 + c] = (__bf16)ref[((size_t)(b * kC + c)) * kHW + sp];
  }
  o[128] = (__bf16)fx;
  o[129] = (__bf16)fy;
  for (int c = 130; c < 160; ++c) o[c] = (__bf16)0.f;
}

// NCHW f32 -> padded NHWC bf16 (stride + channel offset, to build concats)
__global__ __launch_bounds__(256) void nchw_to_nhwc(const float* __restrict__ in,
                                                    __bf16* __restrict__ out,
                                                    int Cc, int outStride, int chOff) {
  long i = (long)blockIdx.x * 256 + threadIdx.x;
  long tot = (long)kB * Cc * kHW;
  if (i >= tot) return;
  int sp = (int)(i % kHW);
  long r = i / kHW;
  int c  = (int)(r % Cc);
  int b  = (int)(r / Cc);
  int y = sp >> 8, x = sp & (kW - 1);
  out[((size_t)(b * kPH + y + 1) * kPW + x + 1) * outStride + chOff + c] =
      (__bf16)in[i];
}

// ---------------------------------------------------------------------------
// 3x3 conv as implicit GEMM on WMMA.
//   in : padded NHWC bf16, row stride Cin (multiple of 32)
//   wr : [9][CoutPad][Cin] bf16
// Block = 128 threads (4 waves) covers 64 pixels of one row (4 N-subtiles).
// TDM stages the 3x66xCin activation strip into LDS once; waves consume it
// with ds_load_b128, so the inner loop is LDS + WMMA only (no masking).
//   OPAD=1: out padded NHWC ; OPAD=0: out plain NHWC (offset/mask tensor)
// ---------------------------------------------------------------------------
template <int ACT, int OPAD>
__global__ __launch_bounds__(128) void conv3x3_wmma(
    const __bf16* __restrict__ in, int Cin,
    const __bf16* __restrict__ wr, int CoutPad, int CoutReal,
    const float* __restrict__ bias,
    __bf16* __restrict__ out, int outStride, int outChOff) {
  extern __shared__ __bf16 smem[];            // 3 * 66 * Cin bf16
  int lane  = threadIdx.x & 31;
  int wave  = threadIdx.x >> 5;
  int cout0 = (blockIdx.y * 4 + wave) * 16;

  long pg = (long)blockIdx.x * 64;            // first (unpadded) pixel of block
  int b   = (int)(pg >> 15);
  int sp0 = (int)(pg & (kHW - 1));
  int y   = sp0 >> 8;
  int x0  = sp0 & (kW - 1);                   // 0,64,128,192

  // Stage strip: padded rows y..y+2, padded cols x0..x0+65.
  const __bf16* gsrc = in + ((size_t)(b * kPH + y) * kPW + x0) * Cin;
  const int widthElems = 66 * Cin;
  if (wave == 0) {
#if __has_builtin(__builtin_amdgcn_tensor_load_to_lds)
    tdm_load_strip(gsrc, (unsigned)(size_t)smem, widthElems, 3,
                   (long)kPW * Cin);
#endif
  }
#if !__has_builtin(__builtin_amdgcn_tensor_load_to_lds)
  for (int rrow = 0; rrow < 3; ++rrow) {
    const u32x4* s = (const u32x4*)(gsrc + (size_t)rrow * kPW * Cin);
    u32x4* d = (u32x4*)(smem + rrow * widthElems);
    int nv = widthElems / 8;
    for (int v = threadIdx.x; v < nv; v += 128) d[v] = s[v];
  }
#endif
  __syncthreads();

  if (cout0 < CoutPad) {
    int n     = lane & 15;     // M row / pixel-in-subtile
    int khalf = lane >> 4;     // K half selector
    v8f acc[4] = {{}, {}, {}, {}};
    for (int t = 0; t < 9; ++t) {
      int dy = t / 3, dx = t % 3;             // padded tap offsets 0..2
      const __bf16* arow =
          wr + (((size_t)t * CoutPad + cout0 + n) * Cin) + khalf * 16;
      const __bf16* srow = smem + (dy * 66) * Cin + khalf * 16;
      for (int k0 = 0; k0 < Cin; k0 += 32) {
        v16bf a = *(const v16bf*)(arow + k0);
        #pragma unroll
        for (int i = 0; i < 4; ++i) {
          int col = i * 16 + n + dx;          // 0..65, always in strip
          v16bf bf = *(const v16bf*)(srow + (size_t)col * Cin + k0);
          acc[i] = __builtin_amdgcn_wmma_f32_16x16x32_bf16(
              false, a, false, bf, (short)0, acc[i], false, false);
        }
      }
    }
    // epilogue: VGPR r holds M = r + 8*khalf, N = lane&15
    #pragma unroll
    for (int i = 0; i < 4; ++i) {
      int xp = x0 + i * 16 + n;
      #pragma unroll
      for (int r = 0; r < 8; ++r) {
        int co = cout0 + r + 8 * khalf;
        float v = acc[i][r];
        if (co < CoutReal) v += bias[co]; else v = 0.f;
        if (ACT) v = (v >= 0.f) ? v : 0.1f * v;
        if (OPAD) {
          out[((size_t)(b * kPH + y + 1) * kPW + xp + 1) * outStride +
              outChOff + co] = (__bf16)v;
        } else {
          out[((size_t)(pg + i * 16 + n)) * outStride + outChOff + co] =
              (__bf16)v;
        }
      }
    }
  }
}

// ---------------------------------------------------------------------------
// Deformable conv: fused bilinear gather + GEMM (M=64, K=576 = 9 taps x 64ch)
//   xin : padded NHWC bf16 (row stride xStride), gathered channels 0..63
//   om  : plain NHWC bf16 stride 224; ch 18g+2k (off_y), 18g+2k+1 (off_x),
//         144+9g+k (mask)
//   OFFMODE 1: off = 20*tanh(om)+flow ;  OFFMODE 2: off = om
//   OUTMODE 0: bf16 padded NHWC ; OUTMODE 2: f32 NCHW into d_out
// ---------------------------------------------------------------------------
__device__ __forceinline__ void gather8(const __bf16* __restrict__ xin, int xStride,
                                        int b, int yy, int xx, float wgt,
                                        int g, float* acc8) {
  if (yy < 0 || yy >= kH || xx < 0 || xx >= kW) return;
  const __bf16* p =
      xin + ((size_t)(b * kPH + yy + 1) * kPW + xx + 1) * xStride + g * 8;
  #pragma unroll
  for (int j = 0; j < 8; ++j) acc8[j] += wgt * (float)p[j];
}

template <int OFFMODE, int OUTMODE>
__global__ __launch_bounds__(128) void dconv_wmma(
    const __bf16* __restrict__ xin, int xStride,
    const __bf16* __restrict__ om,
    const float* __restrict__ flow,
    const __bf16* __restrict__ wr,      // [64][576]
    const float* __restrict__ bias,
    void* __restrict__ out, int outStride) {
  int lane  = threadIdx.x & 31;
  int wave  = threadIdx.x >> 5;
  int cout0 = wave * 16;                 // 4 waves cover Cout = 64

  long pb = (long)blockIdx.x * 16;
  int b   = (int)(pb >> 15);
  int sp0 = (int)(pb & (kHW - 1));
  int y   = sp0 >> 8;
  int x0  = sp0 & (kW - 1);
  int n     = lane & 15;
  int khalf = lane >> 4;
  int xp    = x0 + n;
  int sp    = sp0 + n;

  float fy = 0.f, fx = 0.f;
  if (OFFMODE == 1) {
    fy = flow[((size_t)(b * 2 + 1)) * kHW + sp];
    fx = flow[((size_t)(b * 2 + 0)) * kHW + sp];
  }
  const __bf16* omp = om + (size_t)(pb + n) * 224;
  __builtin_prefetch(omp, 0, 0);

  union Frag { v16bf v; __bf16 e[16]; };
  v8f acc = {};
  for (int k = 0; k < 9; ++k) {
    int ky = k / 3 - 1, kx = k % 3 - 1;
    #pragma unroll
    for (int h = 0; h < 2; ++h) {        // two 32-wide K steps per tap
      Frag bf;
      #pragma unroll
      for (int sub = 0; sub < 2; ++sub) {
        int g = h * 4 + khalf * 2 + sub; // 8-channel offset group
        float oy = (float)omp[18 * g + 2 * k];
        float ox = (float)omp[18 * g + 2 * k + 1];
        float mk = (float)omp[144 + 9 * g + k];
        if (OFFMODE == 1) {
          oy = kMRM * tanhf(oy) + fy;
          ox = kMRM * tanhf(ox) + fx;
        }
        float mval = 1.f / (1.f + __expf(-mk));
        float py = (float)y + (float)ky + oy;
        float px = (float)xp + (float)kx + ox;
        float y0f = floorf(py), x0f = floorf(px);
        int   yi = (int)y0f,    xi = (int)x0f;
        float wy = py - y0f,    wx = px - x0f;
        float a8[8] = {0.f, 0.f, 0.f, 0.f, 0.f, 0.f, 0.f, 0.f};
        gather8(xin, xStride, b, yi,     xi,     (1.f - wy) * (1.f - wx), g, a8);
        gather8(xin, xStride, b, yi,     xi + 1, (1.f - wy) * wx,         g, a8);
        gather8(xin, xStride, b, yi + 1, xi,     wy * (1.f - wx),         g, a8);
        gather8(xin, xStride, b, yi + 1, xi + 1, wy * wx,                 g, a8);
        #pragma unroll
        for (int j = 0; j < 8; ++j) bf.e[sub * 8 + j] = (__bf16)(a8[j] * mval);
      }
      v16bf a = *(const v16bf*)(wr + ((size_t)(cout0 + n)) * 576 +
                                k * 64 + h * 32 + khalf * 16);
      acc = __builtin_amdgcn_wmma_f32_16x16x32_bf16(
          false, a, false, bf.v, (short)0, acc, false, false);
    }
  }

  #pragma unroll
  for (int r = 0; r < 8; ++r) {
    int co = cout0 + r + 8 * khalf;      // < 64
    float v = acc[r] + bias[co];
    if (OUTMODE == 0) {
      ((__bf16*)out)[((size_t)(b * kPH + y + 1) * kPW + xp + 1) * outStride +
                     co] = (__bf16)v;
    } else {
      ((float*)out)[(((size_t)(b * 64 + co)) * kH + y) * kW + xp] = v;
    }
  }
}

// ---------------------------------------------------------------------------
// Host driver
// ---------------------------------------------------------------------------
extern "C" void kernel_launch(void* const* d_in, const int* in_sizes, int n_in,
                              void* d_out, int out_size, void* d_ws, size_t ws_size,
                              hipStream_t stream) {
  (void)in_sizes; (void)n_in; (void)out_size; (void)ws_size;
  const float* nonref = (const float*)d_in[0];
  const float* refp   = (const float*)d_in[1];
  const float* flowp  = (const float*)d_in[2];
  const float* w1  = (const float*)d_in[3];   const float* b1  = (const float*)d_in[4];
  const float* w2  = (const float*)d_in[5];   const float* b2  = (const float*)d_in[6];
  const float* w3  = (const float*)d_in[7];   const float* b3  = (const float*)d_in[8];
  const float* w4  = (const float*)d_in[9];   const float* b4  = (const float*)d_in[10];
  const float* wdc = (const float*)d_in[11];  const float* bdc = (const float*)d_in[12];
  const float* wc1 = (const float*)d_in[13];  const float* bc1 = (const float*)d_in[14];
  const float* wc2 = (const float*)d_in[15];  const float* bc2 = (const float*)d_in[16];
  const float* wom = (const float*)d_in[17];  const float* bom = (const float*)d_in[18];
  const float* wdc2= (const float*)d_in[19];  const float* bdc2= (const float*)d_in[20];

  char* base = (char*)d_ws;
  size_t off = 0;
  auto alloc = [&](size_t bytes) -> void* {
    void* p = base + off;
    off += (bytes + 255) & ~(size_t)255;
    return p;
  };
  const size_t padPix = (size_t)kB * kPP;
  __bf16* hin  = (__bf16*)alloc(padPix * 160 * 2);   // conv1 input (padded)
  __bf16* actA = (__bf16*)alloc(padPix * 64 * 2);    // padded
  __bf16* actB = (__bf16*)alloc(padPix * 64 * 2);    // padded
  __bf16* omB  = (__bf16*)alloc((size_t)kP * 224 * 2); // offsets/mask (plain)
  __bf16* nrN  = (__bf16*)alloc(padPix * 64 * 2);    // non_ref NHWC (padded)
  __bf16* h2   = (__bf16*)alloc(padPix * 128 * 2);   // [feat | ref] (padded)
  __bf16* wr1  = (__bf16*)alloc((size_t)9 * 64 * 160 * 2);
  __bf16* wr2  = (__bf16*)alloc((size_t)9 * 64 * 64 * 2);
  __bf16* wr3  = (__bf16*)alloc((size_t)9 * 64 * 64 * 2);
  __bf16* wr4  = (__bf16*)alloc((size_t)9 * 224 * 64 * 2);
  __bf16* wrc1 = (__bf16*)alloc((size_t)9 * 64 * 128 * 2);
  __bf16* wrc2 = (__bf16*)alloc((size_t)9 * 64 * 64 * 2);
  __bf16* wrom = (__bf16*)alloc((size_t)9 * 224 * 64 * 2);
  __bf16* wrdc = (__bf16*)alloc((size_t)64 * 576 * 2);
  __bf16* wrdc2= (__bf16*)alloc((size_t)64 * 576 * 2);

  auto cdiv = [](long a, long b) { return (int)((a + b - 1) / b); };

  // ---- weight reorders ----
  prep_conv_w<<<cdiv(9L*64*160, 256), 256, 0, stream>>>(w1,  wr1,  64, 64, 130, 160);
  prep_conv_w<<<cdiv(9L*64*64,  256), 256, 0, stream>>>(w2,  wr2,  64, 64, 64, 64);
  prep_conv_w<<<cdiv(9L*64*64,  256), 256, 0, stream>>>(w3,  wr3,  64, 64, 64, 64);
  prep_conv_w<<<cdiv(9L*224*64, 256), 256, 0, stream>>>(w4,  wr4,  216, 224, 64, 64);
  prep_conv_w<<<cdiv(9L*64*128, 256), 256, 0, stream>>>(wc1, wrc1, 64, 64, 128, 128);
  prep_conv_w<<<cdiv(9L*64*64,  256), 256, 0, stream>>>(wc2, wrc2, 64, 64, 64, 64);
  prep_conv_w<<<cdiv(9L*224*64, 256), 256, 0, stream>>>(wom, wrom, 216, 224, 64, 64);
  prep_dc_w<<<cdiv(64L*576, 256), 256, 0, stream>>>(wdc,  wrdc);
  prep_dc_w<<<cdiv(64L*576, 256), 256, 0, stream>>>(wdc2, wrdc2);

  // ---- zero halos, layout conversions, warp ----
  int hb = cdiv((long)kB * kPP, 256);
  zero_halo<<<hb, 256, 0, stream>>>(hin, 160);
  zero_halo<<<hb, 256, 0, stream>>>(actA, 64);
  zero_halo<<<hb, 256, 0, stream>>>(actB, 64);
  zero_halo<<<hb, 256, 0, stream>>>(nrN, 64);
  zero_halo<<<hb, 256, 0, stream>>>(h2, 128);
  warp_build<<<kP / 256, 256, 0, stream>>>(nonref, refp, flowp, hin);
  nchw_to_nhwc<<<cdiv((long)kB*64*kHW, 256), 256, 0, stream>>>(nonref, nrN, 64, 64, 0);
  nchw_to_nhwc<<<cdiv((long)kB*64*kHW, 256), 256, 0, stream>>>(refp, h2, 64, 128, 64);

  dim3 blk(128);
  dim3 g1(kP / 64, 1);    // Cout 64
  dim3 g4(kP / 64, 4);    // Cout 224
  dim3 gd(kP / 16, 1);    // dconv
  auto smem = [](int Cin) { return (size_t)3 * 66 * Cin * 2; };

  // ---- stage 1: convs -> offsets/mask ----
  conv3x3_wmma<1,1><<<g1, blk, smem(160), stream>>>(hin, 160, wr1, 64, 64, b1,
                                                    actA, 64, 0);
  conv3x3_wmma<1,1><<<g1, blk, smem(64),  stream>>>(actA, 64, wr2, 64, 64, b2,
                                                    actB, 64, 0);
  conv3x3_wmma<1,1><<<g1, blk, smem(64),  stream>>>(actB, 64, wr3, 64, 64, b3,
                                                    actA, 64, 0);
  conv3x3_wmma<0,0><<<g4, blk, smem(64),  stream>>>(actA, 64, wr4, 224, 216, b4,
                                                    omB, 224, 0);

  // ---- deform conv 1 (offsets = 20*tanh + flow), feat -> h2[0:64] ----
  dconv_wmma<1, 0><<<gd, blk, 0, stream>>>(nrN, 64, omB, flowp, wrdc, bdc, h2, 128);

  // ---- stage 2: convs on [feat|ref] -> second offsets/mask ----
  conv3x3_wmma<1,1><<<g1, blk, smem(128), stream>>>(h2, 128, wrc1, 64, 64, bc1,
                                                    actA, 64, 0);
  conv3x3_wmma<1,1><<<g1, blk, smem(64),  stream>>>(actA, 64, wrc2, 64, 64, bc2,
                                                    actB, 64, 0);
  conv3x3_wmma<0,0><<<g4, blk, smem(64),  stream>>>(actB, 64, wrom, 224, 216, bom,
                                                    omB, 224, 0);

  // ---- deform conv 2 (raw offsets), write fp32 NCHW output ----
  dconv_wmma<2, 2><<<gd, blk, 0, stream>>>(h2, 128, omB, nullptr, wrdc2, bdc2,
                                           d_out, 0);
}
